// SimpleAttention_64433099374920
// MI455X (gfx1250) — matrix operand order repaired and checked by reading, not compiled, not run on hardware
//
#include <hip/hip_runtime.h>
#include <hip/hip_bf16.h>

typedef __attribute__((ext_vector_type(16))) _Float16 v16h;
typedef __attribute__((ext_vector_type(8)))  float    v8f;

#define HW    4096
#define CH    64
#define BATCH 4

// LDS row strides (halfs): 16B-aligned rows (=> ds_load_b128 usable) AND
// bank-conflict-free for fragment access (36n mod 64 / 20m mod 64 distinct, n,m<16)
#define KLD 72   // K/V tile row stride
#define PLD 40   // P tile row stride

#define WMMA(d, a, b) \
  d = __builtin_amdgcn_wmma_f32_16x16x32_f16(false, a, false, b, (short)0, d, false, false)

// ---------------- WMMA fragment builders (CDNA5 ISA §7.12.2) ----------------
// 16-bit A matrix 16x32 (MxK): lane L holds row M=L%16; half h=L/16:
//   VGPR v -> K = k0 + 8*h + (v<4 ? 0 : 16) + 2*(v&3)   (pairs contiguous -> dword loads)
__device__ __forceinline__ v16h load_A16x32(const _Float16* base, int ld, int k0) {
  int lane = threadIdx.x & 31;
  int h = lane >> 4, m = lane & 15;
  union { v16h v; unsigned u[8]; } a;
  const _Float16* row = base + m * ld + k0 + 8 * h;
#pragma unroll
  for (int i = 0; i < 8; ++i) {
    int kk = ((i < 4) ? 0 : 16) + 2 * (i & 3);
    a.u[i] = *(const unsigned*)(row + kk);
  }
  return a.v;
}

// 16-bit B matrix 32x16 (KxN): lane L holds column N=L%16; half h=L/16:
//   VGPR v -> K = k0 + 8*h + (v<4 ? 0 : 16) + 2*(v&3)  (pairs contiguous along K -> dword loads)
__device__ __forceinline__ v16h load_B32x16(const _Float16* base, int ld, int k0) {
  int lane = threadIdx.x & 31;
  int h = lane >> 4, n = lane & 15;
  union { v16h v; unsigned u[8]; } b;
  const _Float16* col = base + n * ld + k0 + 8 * h;
#pragma unroll
  for (int i = 0; i < 8; ++i) b.u[i] = *(const unsigned*)(col + ((i < 4) ? 0 : 16) + 2 * (i & 3));
  return b.v;
}

// ---------------- DPP butterfly reductions over 16-lane rows (VALU only, no LDS) -------
template <int CTRL>
__device__ __forceinline__ float fmax_bfly(float x) {
  union { float f; int i; } a, b;
  a.f = x;
  b.i = __builtin_amdgcn_update_dpp(a.i, a.i, CTRL, 0xF, 0xF, true);
  return fmaxf(a.f, b.f);
}
template <int CTRL>
__device__ __forceinline__ float fadd_bfly(float x) {
  union { float f; int i; } a, b;
  a.f = x;
  b.i = __builtin_amdgcn_update_dpp(a.i, a.i, CTRL, 0xF, 0xF, true);
  return a.f + b.f;
}
__device__ __forceinline__ float rowmax16(float x) {
  x = fmax_bfly<0xB1>(x);   // quad_perm [1,0,3,2] : xor 1
  x = fmax_bfly<0x4E>(x);   // quad_perm [2,3,0,1] : xor 2
  x = fmax_bfly<0x141>(x);  // row_half_mirror     : xor 4 (after quad reduce)
  x = fmax_bfly<0x140>(x);  // row_mirror          : xor 8 (after half reduce)
  return x;
}
__device__ __forceinline__ float rowsum16(float x) {
  x = fadd_bfly<0xB1>(x);
  x = fadd_bfly<0x4E>(x);
  x = fadd_bfly<0x141>(x);
  x = fadd_bfly<0x140>(x);
  return x;
}

// ---------------- 1) GroupNorm: one block per (batch, group) ----------------
__global__ void gn_kernel(const float* __restrict__ x, const float* __restrict__ gamma,
                          const float* __restrict__ beta, float* __restrict__ xn) {
  int bg = blockIdx.x;              // 0..31
  int b = bg >> 3, g = bg & 7;
  const int CPG = CH / 8;
  const int N = CPG * HW;           // 32768
  const float* src = x  + ((size_t)b * CH + g * CPG) * HW;
  float*       dst = xn + ((size_t)b * CH + g * CPG) * HW;
  __shared__ float red[256], red2[256];
  float s = 0.f, s2 = 0.f;
  for (int i = threadIdx.x; i < N; i += 256) { float t = src[i]; s += t; s2 += t * t; }
  red[threadIdx.x] = s; red2[threadIdx.x] = s2;
  __syncthreads();
  for (int off = 128; off > 0; off >>= 1) {
    if ((int)threadIdx.x < off) {
      red[threadIdx.x]  += red[threadIdx.x + off];
      red2[threadIdx.x] += red2[threadIdx.x + off];
    }
    __syncthreads();
  }
  float mu  = red[0] / (float)N;
  float var = red2[0] / (float)N - mu * mu;
  float inv = rsqrtf(var + 1e-5f);
  for (int i = threadIdx.x; i < N; i += 256) {
    int c = g * CPG + i / HW;
    dst[i] = (src[i] - mu) * inv * gamma[c] + beta[c];
  }
}

// ---------------- 2) QKV 1x1 conv -> token-major f16 q/k/v ----------------
__global__ void qkv_kernel(const float* __restrict__ xn, const float* __restrict__ w,
                           const float* __restrict__ bias, _Float16* __restrict__ q,
                           _Float16* __restrict__ k, _Float16* __restrict__ v) {
  __shared__ float wl[192 * 64];
  __shared__ float xt[64 * 64];   // [c][t]
  int tt = blockIdx.x, b = blockIdx.y, tid = threadIdx.x;
  for (int i = tid; i < 192 * 64; i += 256) wl[i] = w[i];
  const float* xb = xn + (size_t)b * CH * HW + tt * 64;
  for (int i = tid; i < 64 * 64; i += 256) {
    int c = i >> 6, t = i & 63;
    xt[c * 64 + t] = xb[(size_t)c * HW + t];
  }
  __syncthreads();
  int t = tid & 63, gq = tid >> 6;
  size_t tok = (size_t)b * HW + tt * 64 + t;
  for (int oi = 0; oi < 48; ++oi) {
    int o = gq + 4 * oi;
    float acc = bias[o];
#pragma unroll 8
    for (int c = 0; c < 64; ++c) acc += wl[o * 64 + c] * xt[c * 64 + t];
    _Float16 hv = (_Float16)acc;
    int oc = o & 63;
    if (o < 64)       q[tok * 64 + oc] = hv;
    else if (o < 128) k[tok * 64 + oc] = hv;
    else              v[tok * 64 + oc] = hv;
  }
}

// ---------------- 3) Flash attention ----------------
// 128 threads = 4 waves; each wave owns TWO 16-row Q sets (32 query rows) so every
// K/V B-fragment loaded from LDS feeds two WMMAs. 64-key tiles, double-buffered LDS
// (one barrier per tile), 32 v_wmma_f32_16x16x32_f16 per staged tile per wave.
__global__ void __launch_bounds__(128) fa_kernel(const _Float16* __restrict__ q,
                                                 const _Float16* __restrict__ k,
                                                 const _Float16* __restrict__ v,
                                                 float* __restrict__ out) {
  __shared__ __align__(16) _Float16 Kl[2][64 * KLD];        // [buf][key][c]
  __shared__ __align__(16) _Float16 Vt[2][64 * KLD];        // [buf][c][key]
  __shared__ __align__(16) _Float16 Pl[4][2][16 * PLD];     // [wave][set][m][k]
  int b = blockIdx.y;
  int qt = blockIdx.x;                  // 0..31 query tiles of 128
  int tid  = threadIdx.x;
  int wave = tid >> 5;
  int lane = tid & 31;
  int hh = lane >> 4, nn = lane & 15;
  int qrow0 = qt * 128 + wave * 32;     // 32 rows per wave (2 sets of 16)

  v16h A[2][2];
#pragma unroll
  for (int s = 0; s < 2; ++s) {
    const _Float16* qb = q + ((size_t)b * HW + qrow0 + s * 16) * CH;
    A[s][0] = load_A16x32(qb, CH, 0);
    A[s][1] = load_A16x32(qb, CH, 32);
  }

  v8f O[2][4] = {};
  float mst[2][8], lst[2][8];
#pragma unroll
  for (int s = 0; s < 2; ++s)
#pragma unroll
    for (int r = 0; r < 8; ++r) { mst[s][r] = -1e30f; lst[s][r] = 0.f; }
  const float sc2 = 0.125f * 1.4426950408889634f;   // (1/sqrt(64)) * log2(e)

  const unsigned* ksrc = (const unsigned*)(k + (size_t)b * HW * CH);
  const unsigned* vsrc = (const unsigned*)(v + (size_t)b * HW * CH);

  // -------- stage tile 0 into buffer 0
  {
    unsigned* Klu = (unsigned*)Kl[0];
    unsigned* Vtu = (unsigned*)Vt[0];
#pragma unroll
    for (int i = 0; i < 16; ++i) {
      int idx = tid + i * 128;
      Klu[(idx >> 5) * (KLD / 2) + (idx & 31)] = ksrc[idx];
    }
#pragma unroll
    for (int i = 0; i < 8; ++i) {
      int p = tid + i * 128;
      int cp = p & 31, kp = p >> 5;
      unsigned a = vsrc[(2 * kp) * 32 + cp];
      unsigned bb = vsrc[(2 * kp + 1) * 32 + cp];
      Vtu[(2 * cp) * (KLD / 2) + kp]     = (a & 0xffffu) | (bb << 16);
      Vtu[(2 * cp + 1) * (KLD / 2) + kp] = (a >> 16) | (bb & 0xffff0000u);
    }
  }
  __syncthreads();

  for (int kb = 0; kb < HW; kb += 64) {
    int cur = (kb >> 6) & 1;
    // -------- prefetch next tile into the other buffer (overlaps with compute)
    if (kb + 64 < HW) {
      const unsigned* kn = ksrc + (kb + 64) * 32;
      const unsigned* vn = vsrc + (kb + 64) * 32;
      unsigned* Klu = (unsigned*)Kl[cur ^ 1];
      unsigned* Vtu = (unsigned*)Vt[cur ^ 1];
#pragma unroll
      for (int i = 0; i < 16; ++i) {
        int idx = tid + i * 128;
        Klu[(idx >> 5) * (KLD / 2) + (idx & 31)] = kn[idx];
      }
#pragma unroll
      for (int i = 0; i < 8; ++i) {
        int p = tid + i * 128;
        int cp = p & 31, kp = p >> 5;
        unsigned a = vn[(2 * kp) * 32 + cp];
        unsigned bb = vn[(2 * kp + 1) * 32 + cp];
        Vtu[(2 * cp) * (KLD / 2) + kp]     = (a & 0xffffu) | (bb << 16);
        Vtu[(2 * cp + 1) * (KLD / 2) + kp] = (a >> 16) | (bb & 0xffff0000u);
      }
    }

    // -------- compute: two 32-key halves; each B fragment feeds 2 WMMAs (2 Q sets)
#pragma unroll
    for (int half = 0; half < 2; ++half) {
      const _Float16* Kb0 = Kl[cur] + (half * 32) * KLD;   // key subtile 0
      const _Float16* Kb1 = Kb0 + 16 * KLD;                // key subtile 1
      v8f S[2][2] = {};
      v16h bk;
      bk = load_B32x16(Kb0, KLD, 32);
      WMMA(S[0][0], A[0][1], bk); WMMA(S[1][0], A[1][1], bk);
      bk = load_B32x16(Kb0, KLD, 0);
      WMMA(S[0][0], A[0][0], bk); WMMA(S[1][0], A[1][0], bk);
      bk = load_B32x16(Kb1, KLD, 32);
      WMMA(S[0][1], A[0][1], bk); WMMA(S[1][1], A[1][1], bk);
      bk = load_B32x16(Kb1, KLD, 0);
      WMMA(S[0][1], A[0][0], bk); WMMA(S[1][1], A[1][0], bk);

      // online softmax in exp2 domain (per Q set). C layout: (M=r+8*(lane/16), N=lane%16)
      // Row max reduced now via DPP; row sums stay per-lane partial (cf lane-uniform).
#pragma unroll
      for (int s = 0; s < 2; ++s) {
        _Float16* pw = Pl[wave][s];
#pragma unroll
        for (int r = 0; r < 8; ++r) {
          float rm = rowmax16(fmaxf(S[s][0][r], S[s][1][r]) * sc2);
          float mn = fmaxf(mst[s][r], rm);
          float cf = exp2f(mst[s][r] - mn);
          mst[s][r] = mn;
          float p0 = exp2f(S[s][0][r] * sc2 - mn);
          float p1 = exp2f(S[s][1][r] * sc2 - mn);
          lst[s][r] = lst[s][r] * cf + p0 + p1;
          O[s][0][r] *= cf; O[s][1][r] *= cf; O[s][2][r] *= cf; O[s][3][r] *= cf;
          int row = r + 8 * hh;
          pw[row * PLD + nn]      = (_Float16)p0;
          pw[row * PLD + 16 + nn] = (_Float16)p1;
        }
      }

      // O += P(16x32) * V(32x64): each V fragment feeds both Q sets
      v16h ap0 = load_A16x32(Pl[wave][0], PLD, 0);
      v16h ap1 = load_A16x32(Pl[wave][1], PLD, 0);
      v16h bv;
#pragma unroll
      for (int nt = 0; nt < 4; ++nt) {
        bv = load_B32x16(Vt[cur] + nt * 16 * KLD, KLD, half * 32);
        WMMA(O[0][nt], ap0, bv);
        WMMA(O[1][nt], ap1, bv);
      }
    }
    __syncthreads();
  }

  // -------- epilogue: final cross-lane row sums, normalize, write token-major f32
#pragma unroll
  for (int s = 0; s < 2; ++s) {
#pragma unroll
    for (int r = 0; r < 8; ++r) {
      float invl = 1.f / rowsum16(lst[s][r]);
      int row = qrow0 + s * 16 + r + 8 * hh;
      float* o = out + ((size_t)b * HW + row) * CH;
      o[nn]      = O[s][0][r] * invl;
      o[16 + nn] = O[s][1][r] * invl;
      o[32 + nn] = O[s][2][r] * invl;
      o[48 + nn] = O[s][3][r] * invl;
    }
  }
}

// ---------------- 4) Output projection + residual ----------------
__global__ void proj_kernel(const float* __restrict__ ao, const float* __restrict__ w,
                            const float* __restrict__ bias, const float* __restrict__ x,
                            float* __restrict__ out) {
  __shared__ float wl[64 * 64];
  __shared__ float at[64 * 64];   // [t][c]
  int tt = blockIdx.x, b = blockIdx.y, tid = threadIdx.x;
  for (int i = tid; i < 4096; i += 256) wl[i] = w[i];
  const float* as = ao + ((size_t)b * HW + tt * 64) * CH;
  for (int i = tid; i < 4096; i += 256) at[i] = as[i];
  __syncthreads();
  int t = tid & 63, gq = tid >> 6;
  for (int oi = 0; oi < 16; ++oi) {
    int o = gq + 4 * oi;
    float acc = bias[o];
#pragma unroll 8
    for (int c = 0; c < 64; ++c) acc += wl[o * 64 + c] * at[t * 64 + c];
    size_t idx = ((size_t)b * CH + o) * HW + tt * 64 + t;
    out[idx] = acc + x[idx];
  }
}

// ---------------- launch ----------------
extern "C" void kernel_launch(void* const* d_in, const int* in_sizes, int n_in,
                              void* d_out, int out_size, void* d_ws, size_t ws_size,
                              hipStream_t stream) {
  const float* x      = (const float*)d_in[0];
  const float* gamma  = (const float*)d_in[1];
  const float* beta   = (const float*)d_in[2];
  const float* w_qkv  = (const float*)d_in[3];
  const float* b_qkv  = (const float*)d_in[4];
  const float* w_proj = (const float*)d_in[5];
  const float* b_proj = (const float*)d_in[6];
  float* out = (float*)d_out;

  const size_t NTOK = (size_t)BATCH * HW;        // 16384
  float*    xn = (float*)d_ws;                   // 4 MB
  _Float16* q  = (_Float16*)(xn + NTOK * CH);    // 2 MB each
  _Float16* k  = q + NTOK * CH;
  _Float16* v  = k + NTOK * CH;
  float*    ao = (float*)(v + NTOK * CH);        // 4 MB

  gn_kernel  <<<32,          256, 0, stream>>>(x, gamma, beta, xn);
  qkv_kernel <<<dim3(64, 4), 256, 0, stream>>>(xn, w_qkv, b_qkv, q, k, v);
  fa_kernel  <<<dim3(32, 4), 128, 0, stream>>>(q, k, v, ao);
  proj_kernel<<<dim3(64, 4), 256, 0, stream>>>(ao, w_proj, b_proj, x, out);
}
